// ResNet50GRAPHPOOLPARTHyper_40862318854446
// MI455X (gfx1250) — compile-verified, matrix operand order checked by reading, not run on hardware
//
#include <hip/hip_runtime.h>
#include <hip/hip_bf16.h>

typedef _Float16 v8h  __attribute__((ext_vector_type(8)));
typedef _Float16 v16h __attribute__((ext_vector_type(16)));
typedef float    v8f  __attribute__((ext_vector_type(8)));
typedef float    v2f  __attribute__((ext_vector_type(2)));

#define D      2048
#define NTOK   8192      // b*N = 16*512
#define NB     64        // blocks per sequence
#define P      8
#define CAND   56        // 7*P
#define COUT   1024      // 2*dout
#define DOUT   512
// main GEMM tiling
#define TM     64
#define TN     128
#define KB     64
#define LSA    72        // 64 + 8 halves pad (144B rows, 16B aligned)
// sim GEMM tiling
#define SKB    64
#define SLS    68        // 64 + 4 floats pad (272B rows, 16B aligned)

__device__ inline v8h cvt8(float4 a, float4 b) {
    v8h r;
    r[0] = (_Float16)a.x; r[1] = (_Float16)a.y; r[2] = (_Float16)a.z; r[3] = (_Float16)a.w;
    r[4] = (_Float16)b.x; r[5] = (_Float16)b.y; r[6] = (_Float16)b.z; r[7] = (_Float16)b.w;
    return r;
}

// ---------------- 1) per-token reciprocal L2 norm of x ----------------
__global__ __launch_bounds__(256) void rnorm_kernel(const float* __restrict__ x,
                                                    float* __restrict__ rn) {
    __shared__ float red[256];
    const int tok = blockIdx.x;
    const float* p = x + (long)tok * D;
    float ss = 0.f;
    for (int i = threadIdx.x; i < D; i += 256) { float v = p[i]; ss += v * v; }
    red[threadIdx.x] = ss; __syncthreads();
    for (int s = 128; s > 0; s >>= 1) {
        if (threadIdx.x < s) red[threadIdx.x] += red[threadIdx.x + s];
        __syncthreads();
    }
    if (threadIdx.x == 0) rn[tok] = 1.0f / fmaxf(sqrtf(red[0]), 1e-12f);
}

// ---------------- 2) fp32-WMMA cosine sims + top-3 --------------------
// Per (n, batch): sim(16x64) = Q(16x2048) x K^T(2048x64) via v_wmma_f32_16x16x4_f32.
// Rows 8..15 of Q duplicate 0..7; cols 56..63 clamp to 55; both ignored at selection.
__device__ inline int cand_tok(int batch, int n, int kk) {
    const int ckk = kk < CAND ? kk : CAND - 1;
    int blk = n + (ckk >> 3) - 3;
    blk = blk < 0 ? 0 : (blk >= NB ? NB - 1 : blk);
    return batch * 512 + blk * P + (ckk & 7);
}

__global__ __launch_bounds__(256) void topk_kernel(const float* __restrict__ x,
                                                   const float* __restrict__ rn,
                                                   int* __restrict__ nbr) {
    __shared__ __align__(16) float Qs[16 * SLS];
    __shared__ __align__(16) float Ks[64 * SLS];
    __shared__ float simacc[2 * 16 * 64];

    const int n = blockIdx.x;          // 0..63
    const int batch = blockIdx.y;      // 0..15
    const int tid = threadIdx.x;
    const int tqbase = batch * 512 + n * P;

    // staging maps (fixed per thread)
    const int qr = tid >> 4, qc = (tid & 15) * 4;            // Q: 16 rows x 64 cols
    const int kr = tid >> 2, kcs = (tid & 3) * 16;           // K: 64 rows x 64 cols
    const float* qp = x + (long)(tqbase + (qr & 7)) * D + qc;
    const float* kp = x + (long)cand_tok(batch, n, kr) * D + kcs;

    // wave tiling: nt = N-tile (16 cols), kh = K-half
    const int lane = tid & 31;
    const int w = tid >> 5;
    const int nt = w & 3;
    const int kh = w >> 2;
    const int lcol = lane & 15;
    const int klo = (lane & 16) ? 2 : 0;

    v8f acc;
    for (int i = 0; i < 8; i++) acc[i] = 0.f;

    for (int kk0 = 0; kk0 < D; kk0 += SKB) {
        __syncthreads();
        *(float4*)&Qs[qr * SLS + qc] = *(const float4*)(qp + kk0);
        #pragma unroll
        for (int u = 0; u < 4; u++)
            *(float4*)&Ks[kr * SLS + kcs + 4 * u] = *(const float4*)(kp + kk0 + 4 * u);
        __syncthreads();

        #pragma unroll
        for (int ks = 0; ks < 8; ks++) {
            const int kofs = kh * 32 + ks * 4;
            v2f a = *(const v2f*)&Qs[lcol * SLS + kofs + klo];
            v2f b = *(const v2f*)&Ks[(nt * 16 + lcol) * SLS + kofs + klo];
            acc = __builtin_amdgcn_wmma_f32_16x16x4_f32(false, a, false, b,
                                                        (short)0, acc, false, false);
        }
    }
    __syncthreads();
    #pragma unroll
    for (int i = 0; i < 8; i++) {
        const int row = i + ((lane & 16) ? 8 : 0);
        simacc[kh * 1024 + row * 64 + nt * 16 + lcol] = acc[i];
    }
    __syncthreads();

    if (tid < P) {
        const int q = tid;
        const int tq = tqbase + q;
        const float rq = rn[tq];
        float m1 = -__builtin_inff(), m2 = m1, m3 = m1;
        int i1 = 0, i2 = 0, i3 = 0;
        for (int kk = 0; kk < CAND; kk++) {
            const int blk = n + (kk >> 3) - 3;
            float s;
            if (blk < 0 || blk >= NB) s = -__builtin_inff();
            else s = (simacc[q * 64 + kk] + simacc[1024 + q * 64 + kk]) * rq
                     * rn[batch * 512 + blk * P + (kk & 7)];
            if (s > m1)      { m3 = m2; i3 = i2; m2 = m1; i2 = i1; m1 = s; i1 = kk; }
            else if (s > m2) { m3 = m2; i3 = i2; m2 = s; i2 = kk; }
            else if (s > m3) { m3 = s; i3 = kk; }
        }
        (void)i1;
        nbr[tq * 2 + 0] = cand_tok(batch, n, i2);   // rank-1 (top-1 excluded)
        nbr[tq * 2 + 1] = cand_tok(batch, n, i3);   // rank-2
    }
}

// ---------------- 3) f16-WMMA GEMM: H = [x@Wx^T+bx | xn@Wn^T+bn] ------
// 64Mx128N block tile, KB=64; 8 waves each do a 32x32 register tile
// (2 A-frags x 2 B-frags = 4 wmma per 32-K step). Global loads for the next
// K-slab are issued before the WMMAs of the current one (latency hiding).
__global__ __launch_bounds__(256) void gemm_kernel(const float* __restrict__ x,
                                                   const float* __restrict__ Wx,
                                                   const float* __restrict__ bx,
                                                   const float* __restrict__ Wn,
                                                   const float* __restrict__ bn,
                                                   const int* __restrict__ nbr,
                                                   float* __restrict__ out) {
    __shared__ __align__(16) _Float16 As[TM * LSA];
    __shared__ __align__(16) _Float16 Bs[TN * LSA];

    const int tid = threadIdx.x;
    const int z = blockIdx.z;                       // 0: x/Wx, 1: x_neib/Wn
    const int M0 = blockIdx.x * TM;
    const int N0 = blockIdx.y * TN;
    const float* W = z ? Wn : Wx;
    const float* bias = z ? bn : bx;
    const int colbase = z ? DOUT : 0;

    // staging maps: A 64x64 (4 thr/row, 16 f32 each), B 128x64 (2 thr/row, 32 f32)
    const int ar = tid >> 2, acs = (tid & 3) * 16;
    const int br = tid >> 1, bcs = (tid & 1) * 32;
    long arow0, arow1;
    if (z == 0) { arow0 = arow1 = (long)(M0 + ar); }
    else        { arow0 = nbr[(M0 + ar) * 2 + 0]; arow1 = nbr[(M0 + ar) * 2 + 1]; }
    const float* a0p = x + arow0 * D + acs;
    const float* a1p = x + arow1 * D + acs;
    const float* bp  = W + (long)(N0 + br) * D + bcs;

    // wave tiling: 2(M) x 4(N) waves, each 32x32
    const int lane = tid & 31;
    const int w = tid >> 5;
    const int wm = (w >> 2) * 32;
    const int wn = (w & 3) * 32;
    const int lcol = lane & 15;
    const int hi = (lane & 16) ? 8 : 0;

    v8f acc00, acc01, acc10, acc11;
    const float bv0 = bias[N0 + wn + lcol];
    const float bv1 = bias[N0 + wn + 16 + lcol];
    for (int i = 0; i < 8; i++) { acc00[i] = bv0; acc01[i] = bv1; acc10[i] = bv0; acc11[i] = bv1; }

    float4 aR[4], aS[4], bR[8];
    // prologue: fetch K-slab 0
    #pragma unroll
    for (int u = 0; u < 4; u++) aR[u] = *(const float4*)(a0p + 4 * u);
    if (z) {
        #pragma unroll
        for (int u = 0; u < 4; u++) aS[u] = *(const float4*)(a1p + 4 * u);
    }
    #pragma unroll
    for (int u = 0; u < 8; u++) bR[u] = *(const float4*)(bp + 4 * u);

    for (int kk = 0; kk < D; kk += KB) {
        if (z) {
            #pragma unroll
            for (int u = 0; u < 4; u++) {
                aR[u].x = 0.5f * (aR[u].x + aS[u].x); aR[u].y = 0.5f * (aR[u].y + aS[u].y);
                aR[u].z = 0.5f * (aR[u].z + aS[u].z); aR[u].w = 0.5f * (aR[u].w + aS[u].w);
            }
        }
        __syncthreads();
        *(v8h*)&As[ar * LSA + acs]     = cvt8(aR[0], aR[1]);
        *(v8h*)&As[ar * LSA + acs + 8] = cvt8(aR[2], aR[3]);
        #pragma unroll
        for (int u = 0; u < 4; u++)
            *(v8h*)&Bs[br * LSA + bcs + 8 * u] = cvt8(bR[2 * u], bR[2 * u + 1]);
        __syncthreads();

        // prefetch next K-slab while WMMAs execute
        if (kk + KB < D) {
            #pragma unroll
            for (int u = 0; u < 4; u++) aR[u] = *(const float4*)(a0p + kk + KB + 4 * u);
            if (z) {
                #pragma unroll
                for (int u = 0; u < 4; u++) aS[u] = *(const float4*)(a1p + kk + KB + 4 * u);
            }
            #pragma unroll
            for (int u = 0; u < 8; u++) bR[u] = *(const float4*)(bp + kk + KB + 4 * u);
        }

        #pragma unroll
        for (int ks = 0; ks < KB; ks += 32) {
            v8h t0, t1;
            t0 = *(const v8h*)&As[(wm + lcol) * LSA + ks + hi];
            t1 = *(const v8h*)&As[(wm + lcol) * LSA + ks + hi + 16];
            v16h a0 = __builtin_shufflevector(t0, t1, 0,1,2,3,4,5,6,7,8,9,10,11,12,13,14,15);
            t0 = *(const v8h*)&As[(wm + 16 + lcol) * LSA + ks + hi];
            t1 = *(const v8h*)&As[(wm + 16 + lcol) * LSA + ks + hi + 16];
            v16h a1 = __builtin_shufflevector(t0, t1, 0,1,2,3,4,5,6,7,8,9,10,11,12,13,14,15);
            t0 = *(const v8h*)&Bs[(wn + lcol) * LSA + ks + hi];
            t1 = *(const v8h*)&Bs[(wn + lcol) * LSA + ks + hi + 16];
            v16h b0 = __builtin_shufflevector(t0, t1, 0,1,2,3,4,5,6,7,8,9,10,11,12,13,14,15);
            t0 = *(const v8h*)&Bs[(wn + 16 + lcol) * LSA + ks + hi];
            t1 = *(const v8h*)&Bs[(wn + 16 + lcol) * LSA + ks + hi + 16];
            v16h b1 = __builtin_shufflevector(t0, t1, 0,1,2,3,4,5,6,7,8,9,10,11,12,13,14,15);

            acc00 = __builtin_amdgcn_wmma_f32_16x16x32_f16(false, a0, false, b0, (short)0, acc00, false, false);
            acc01 = __builtin_amdgcn_wmma_f32_16x16x32_f16(false, a0, false, b1, (short)0, acc01, false, false);
            acc10 = __builtin_amdgcn_wmma_f32_16x16x32_f16(false, a1, false, b0, (short)0, acc10, false, false);
            acc11 = __builtin_amdgcn_wmma_f32_16x16x32_f16(false, a1, false, b1, (short)0, acc11, false, false);
        }
    }

    #pragma unroll
    for (int i = 0; i < 8; i++) {
        const long r0 = (long)(M0 + wm + i + hi) * COUT + colbase;
        const long r1 = (long)(M0 + wm + 16 + i + hi) * COUT + colbase;
        const int c0 = N0 + wn + lcol;
        out[r0 + c0]      = acc00[i];
        out[r0 + c0 + 16] = acc01[i];
        out[r1 + c0]      = acc10[i];
        out[r1 + c0 + 16] = acc11[i];
    }
}

// ---------------- 4) per-row reciprocal L2 norm of H --------------------
__global__ __launch_bounds__(256) void scale_kernel(const float* __restrict__ h,
                                                    float* __restrict__ scale) {
    __shared__ float red[256];
    const int tok = blockIdx.x;
    const float* p = h + (long)tok * COUT;
    float ss = 0.f;
    for (int g = 0; g < 4; g++) { float v = p[threadIdx.x + g * 256]; ss += v * v; }
    red[threadIdx.x] = ss; __syncthreads();
    for (int s = 128; s > 0; s >>= 1) {
        if (threadIdx.x < s) red[threadIdx.x] += red[threadIdx.x + s];
        __syncthreads();
    }
    if (threadIdx.x == 0) scale[tok] = 1.0f / fmaxf(sqrtf(red[0]), 1e-12f);
}

// ---------------- 5) deterministic partial BN stats ---------------------
__global__ __launch_bounds__(256) void partial_kernel(const float* __restrict__ h,
                                                      const float* __restrict__ scale,
                                                      float* __restrict__ psum,
                                                      float* __restrict__ psq) {
    const int bb = blockIdx.x;
    const int t = threadIdx.x;
    float s0[4] = {0,0,0,0}, q0[4] = {0,0,0,0};
    for (int tk = 0; tk < 32; tk++) {
        const int tok = bb * 32 + tk;
        const float sc = scale[tok];
        for (int g = 0; g < 4; g++) {
            float v = h[(long)tok * COUT + t + g * 256] * sc;
            v = fmaxf(v, 0.f);
            s0[g] += v; q0[g] += v * v;
        }
    }
    for (int g = 0; g < 4; g++) {
        psum[(long)bb * COUT + t + g * 256] = s0[g];
        psq [(long)bb * COUT + t + g * 256] = q0[g];
    }
}

__global__ __launch_bounds__(256) void stats_kernel(const float* __restrict__ psum,
                                                    const float* __restrict__ psq,
                                                    float* __restrict__ mv) {
    const int c = blockIdx.x * 256 + threadIdx.x;
    float s = 0.f, q = 0.f;
    for (int pb = 0; pb < 256; pb++) {
        s += psum[(long)pb * COUT + c];
        q += psq [(long)pb * COUT + c];
    }
    const float m = s * (1.0f / (float)NTOK);
    const float v = q * (1.0f / (float)NTOK) - m * m;
    mv[c]        = m;
    mv[COUT + c] = 1.0f / sqrtf(v + 1e-5f);
}

// ---------------- 6) apply norm+relu+BN in place -------------------------
__global__ __launch_bounds__(256) void apply_kernel(float* __restrict__ h,
                                                    const float* __restrict__ scale,
                                                    const float* __restrict__ mv,
                                                    const float* __restrict__ gamma,
                                                    const float* __restrict__ beta) {
    const int tok = blockIdx.x;
    const float sc = scale[tok];
    for (int g = 0; g < 4; g++) {
        const int c = threadIdx.x + g * 256;
        float v = fmaxf(h[(long)tok * COUT + c] * sc, 0.f);
        h[(long)tok * COUT + c] = (v - mv[c]) * mv[COUT + c] * gamma[c] + beta[c];
    }
}

extern "C" void kernel_launch(void* const* d_in, const int* in_sizes, int n_in,
                              void* d_out, int out_size, void* d_ws, size_t ws_size,
                              hipStream_t stream) {
    const float* x     = (const float*)d_in[0];
    const float* Wx    = (const float*)d_in[1];
    const float* bx    = (const float*)d_in[2];
    const float* Wn    = (const float*)d_in[3];
    const float* bn    = (const float*)d_in[4];
    const float* gamma = (const float*)d_in[5];
    const float* beta  = (const float*)d_in[6];
    float* out = (float*)d_out;

    char* w = (char*)d_ws;
    float* rn    = (float*)(w);                          // 8192 f32
    float* scale = (float*)(w + 32768);                  // 8192 f32
    int*   nbr   = (int*)  (w + 65536);                  // 8192*2 i32
    float* psum  = (float*)(w + 131072);                 // 256*1024 f32
    float* psq   = (float*)(w + 131072 + 1048576);       // 256*1024 f32
    float* mv    = (float*)(w + 131072 + 2097152);       // 2048 f32

    rnorm_kernel  <<<NTOK, 256, 0, stream>>>(x, rn);
    topk_kernel   <<<dim3(NB, 16), 256, 0, stream>>>(x, rn, nbr);
    gemm_kernel   <<<dim3(NTOK / TM, DOUT / TN, 2), 256, 0, stream>>>(x, Wx, bx, Wn, bn, nbr, out);
    scale_kernel  <<<NTOK, 256, 0, stream>>>(out, scale);
    partial_kernel<<<256, 256, 0, stream>>>(out, scale, psum, psq);
    stats_kernel  <<<4, 256, 0, stream>>>(psum, psq, mv);
    apply_kernel  <<<NTOK, 256, 0, stream>>>(out, scale, mv, gamma, beta);
}
